// GraphAttentionEncoder_9749575762616
// MI455X (gfx1250) — compile-verified, matrix operand order checked by reading, not run on hardware
//
#include <hip/hip_runtime.h>
#include <hip/hip_bf16.h>

// ---------------------------------------------------------------------------
// GraphAttentionEncoder for MI455X (gfx1250, wave32, WMMA).
//
// B=32, N=100, HID=128, HEADS=8, DH=16, L=3.
// Roofline-driven design:
//  * dis_mat [32,100,100,128] f32 = 164 MB dominates.  It is read exactly
//    once and written exactly once per layer: the EdgeTransition GEMM fuses
//    bias + he1[b,j] + he2[b,i] + sigmoid + residual into its epilogue, and
//    ep = dis@We^T is eliminated algebraically (s_e = dis . (We^T a_e),
//    a 128x8 matrix) saving 10.5 GFLOP + 164 MB/layer.
//  * All [M,128]@[128,128] GEMMs use v_wmma_f32_16x16x32_f16 (f16 in, f32
//    accumulate).  Weights are pre-converted to f16 once per launch, and the
//    128x128 f16 weight tile (32 KB) is staged in LDS per workgroup
//    (row-padded to 272 B against bank conflicts) so all 4 waves share one
//    global fetch and B fragments cost two ds_load_b128 with zero v_cvt.
//  * LDS staging uses GLOBAL_LOAD_ASYNC_TO_LDS_B128 (ASYNCcnt-tracked).
//  * d_out's dis region doubles as the working dis buffer across layers.
// ---------------------------------------------------------------------------

typedef __attribute__((ext_vector_type(16))) _Float16 v16h;
typedef __attribute__((ext_vector_type(8)))  _Float16 v8h;
typedef __attribute__((ext_vector_type(8)))  float    v8f;
typedef __attribute__((__vector_size__(4 * sizeof(int)))) int v4i;

#define GB 32
#define GN 100
#define GHID 128
#define GHEADS 8
#define GDH 16
#define WLDS_STRIDE 136   // 128 + 8 halves pad: 272 B row stride in LDS

enum { EPI_NONE = 0, EPI_BIAS = 1, EPI_EDGE = 2 };

__device__ inline v16h frag_combine(v8h lo, v8h hi) {
  return __builtin_shufflevector(lo, hi, 0, 1, 2, 3, 4, 5, 6, 7,
                                 8, 9, 10, 11, 12, 13, 14, 15);
}

// f16 source: elements [0..7] = K kb..kb+7, [8..15] = K kb+16..kb+23.
__device__ inline v16h frag_from_h(const _Float16* p) {
  return frag_combine(*(const v8h*)p, *(const v8h*)(p + 16));
}

// f32 source, converted on the fly (used only for the dis tensor).
__device__ inline v16h frag_from_f(const float* p) {
  v16h f;
#pragma unroll
  for (int t = 0; t < 8; ++t) {
    f[t]     = (_Float16)p[t];
    f[8 + t] = (_Float16)p[16 + t];
  }
  return f;
}

// Y[M,128] = X[M,128] @ W^T (+ epilogue).  W16 is f16 row-major [out,in].
// One wave owns a 16x128 row-block: 8 accumulators, 32 WMMAs, A loaded once
// per K-step, B served from the LDS-staged weight tile.
template <int EPI, bool AF16>
__global__ __launch_bounds__(128) void gemm_h128_wmma(
    const void* __restrict__ Xv, const _Float16* __restrict__ W16,
    const float* __restrict__ bias, float* __restrict__ Y, int M,
    const float* __restrict__ he1, const float* __restrict__ he2,
    const float* __restrict__ disin) {
  __shared__ _Float16 Wl[GHID * WLDS_STRIDE];  // 34816 B

  // ---- stage the 32 KB f16 weight tile into LDS (once per workgroup) ----
#if __has_builtin(__builtin_amdgcn_global_load_async_to_lds_b128)
  for (int t = threadIdx.x; t < 2048; t += 128) {  // 2048 x 16B chunks
    const int row = t >> 4, cp = t & 15;
    __builtin_amdgcn_global_load_async_to_lds_b128(
        (__attribute__((address_space(1))) v4i*)(W16 + row * GHID + cp * 8),
        (__attribute__((address_space(3))) v4i*)(Wl + row * WLDS_STRIDE + cp * 8),
        0, 0);
  }
#if __has_builtin(__builtin_amdgcn_s_wait_asynccnt)
  __builtin_amdgcn_s_wait_asynccnt(0);
#else
  asm volatile("s_wait_asynccnt 0" ::: "memory");
#endif
#else
  {
    uint32_t* dw = (uint32_t*)Wl;
    const uint32_t* sw = (const uint32_t*)W16;
    for (int t = threadIdx.x; t < 8192; t += 128)
      dw[(t >> 6) * (WLDS_STRIDE / 2) + (t & 63)] = sw[t];
  }
#endif
  __syncthreads();

  const int wave   = threadIdx.x >> 5;
  const int lane   = threadIdx.x & 31;
  const int mTile  = blockIdx.x * 4 + wave;
  const int mTiles = M >> 4;
  if (mTile >= mTiles) return;             // wave-uniform; EXEC stays all-1s
  const int hiHalf = (lane & 16) ? 8 : 0;  // K-offset of this lane half
  const int lane15 = lane & 15;
  const int m      = mTile * 16 + lane15;  // A row held by this lane

  const _Float16* xh = nullptr;
  const float*    xf = nullptr;
  if constexpr (AF16) {
    xh = (const _Float16*)Xv + (size_t)m * GHID;
    __builtin_prefetch(xh + 16 * GHID, 0, 1);  // next row-block
  } else {
    xf = (const float*)Xv + (size_t)m * GHID;
    __builtin_prefetch(xf + 16 * GHID, 0, 1);
  }

  v8f acc[8] = {};
#pragma unroll
  for (int ks = 0; ks < 4; ++ks) {
    const int kb = ks * 32 + hiHalf;
    v16h a;
    if constexpr (AF16) a = frag_from_h(xh + kb);
    else                a = frag_from_f(xf + kb);
#pragma unroll
    for (int nt = 0; nt < 8; ++nt) {
      const int  n = nt * 16 + lane15;     // B column held by this lane
      const v16h b = frag_from_h(Wl + n * WLDS_STRIDE + kb);
      acc[nt] = __builtin_amdgcn_wmma_f32_16x16x32_f16(
          false, a, false, b, (short)0, acc[nt], false, false);
    }
  }

  // D layout: VGPR r -> row r (lanes 0-15) / r+8 (lanes 16-31), col = lane&15.
#pragma unroll
  for (int nt = 0; nt < 8; ++nt) {
#pragma unroll
    for (int r = 0; r < 8; ++r) {
      const int    mrow = mTile * 16 + r + hiHalf;
      const int    col  = nt * 16 + lane15;
      const size_t idx  = (size_t)mrow * GHID + col;
      float v = acc[nt][r];
      if (EPI == EPI_BIAS) {
        v += bias[col];
      } else if (EPI == EPI_EDGE) {
        const int j  = mrow % GN;
        const int ii = (mrow / GN) % GN;
        const int bb = mrow / (GN * GN);
        float t = v + bias[col] + he1[(size_t)(bb * GN + j) * GHID + col]
                               + he2[(size_t)(bb * GN + ii) * GHID + col];
        v = 1.0f / (1.0f + __expf(-t)) + disin[idx];
      }
      Y[idx] = v;
    }
  }
}

__global__ void cvt_f2h(const float* __restrict__ in, _Float16* __restrict__ out,
                        int n) {
  const int i = blockIdx.x * 256 + threadIdx.x;
  if (i < n) out[i] = (_Float16)in[i];
}

// awe[c,h] = sum_d We[h*16+d, c] * a_e[d]   (128x8)
__global__ void compute_awe(const float* __restrict__ We,
                            const float* __restrict__ ae,
                            float* __restrict__ awe) {
  const int idx = blockIdx.x * blockDim.x + threadIdx.x;  // 1024
  if (idx >= GHID * GHEADS) return;
  const int c = idx >> 3, hh = idx & 7;
  float s = 0.f;
#pragma unroll
  for (int d = 0; d < GDH; ++d) s += We[(hh * GDH + d) * GHID + c] * ae[d];
  awe[c * GHEADS + hh] = s;
}

__global__ void compute_si_sj(const float* __restrict__ g,
                              const float* __restrict__ ga,  // [48]
                              float* __restrict__ si, float* __restrict__ sj) {
  const int idx = blockIdx.x * blockDim.x + threadIdx.x;  // 25600
  if (idx >= GB * GN * GHEADS) return;
  const int row = idx >> 3, hh = idx & 7;
  const float* gp = g + (size_t)row * GHID + hh * GDH;
  float s0 = 0.f, s1 = 0.f;
#pragma unroll
  for (int d = 0; d < GDH; ++d) {
    const float gv = gp[d];
    s0 += gv * ga[d];
    s1 += gv * ga[GDH + d];
  }
  si[idx] = s0;
  sj[idx] = s1;
}

// Per (b,i): s_e on the fly, leaky-relu, adj mask, softmax over j, aggregate
// att@g, residual into h.  One 128-thread block per (b,i).
__global__ void gat_attention(const float* __restrict__ dis,
                              const int* __restrict__ adj,
                              const float* __restrict__ awe,
                              const float* __restrict__ si,
                              const float* __restrict__ sj,
                              const float* __restrict__ g,
                              float* __restrict__ h) {
  const int b   = blockIdx.x / GN;
  const int i   = blockIdx.x % GN;
  const int tid = threadIdx.x;  // 128
  __shared__ float row[GHID];
  __shared__ float es[GN * GHEADS];
  const size_t baseBI = (size_t)(b * GN + i) * GN;

  for (int j = 0; j < GN; ++j) {
    row[tid] = dis[(baseBI + j) * GHID + tid];
    __syncthreads();
    if (tid < GHEADS) {
      float s = 0.f;
      for (int c = 0; c < GHID; ++c) s += row[c] * awe[c * GHEADS + tid];
      float val = si[(b * GN + i) * GHEADS + tid] +
                  sj[(b * GN + j) * GHEADS + tid] + s;
      val = (val > 0.f) ? val : 0.2f * val;  // leaky_relu slope 0.2
      es[j * GHEADS + tid] = (adj[baseBI + j] == 0) ? -INFINITY : val;
    }
    __syncthreads();
  }
  if (tid < GHEADS) {  // softmax over j per head
    float mx = -INFINITY;
    for (int j = 0; j < GN; ++j) mx = fmaxf(mx, es[j * GHEADS + tid]);
    float sum = 0.f;
    for (int j = 0; j < GN; ++j) {
      const float e = __expf(es[j * GHEADS + tid] - mx);
      es[j * GHEADS + tid] = e;
      sum += e;
    }
    const float inv = 1.0f / sum;
    for (int j = 0; j < GN; ++j) es[j * GHEADS + tid] *= inv;
  }
  __syncthreads();
  const int hh = tid >> 4;
  float accv = 0.f;
  for (int j = 0; j < GN; ++j)
    accv += es[j * GHEADS + hh] * g[(size_t)(b * GN + j) * GHID + tid];
  h[(size_t)(b * GN + i) * GHID + tid] += accv;  // residual
}

// BatchNorm (track_running_stats=False): biased batch stats per channel.
__global__ void bn_stats(const float* __restrict__ X, int M,
                         float* __restrict__ stats) {
  const int c   = blockIdx.x;   // 128 channels
  const int tid = threadIdx.x;  // 256
  float s = 0.f, q = 0.f;
  for (int r = tid; r < M; r += 256) {
    const float x = X[(size_t)r * GHID + c];
    s += x;
    q += x * x;
  }
  __shared__ float ss[256], qq[256];
  ss[tid] = s; qq[tid] = q;
  __syncthreads();
  for (int o = 128; o > 0; o >>= 1) {
    if (tid < o) { ss[tid] += ss[tid + o]; qq[tid] += qq[tid + o]; }
    __syncthreads();
  }
  if (tid == 0) {
    const float mean = ss[0] / (float)M;
    stats[c]        = mean;
    stats[GHID + c] = qq[0] / (float)M - mean * mean;  // biased var
  }
}

__global__ void bn_apply(float* __restrict__ X, long long total,
                         const float* __restrict__ stats,
                         const float* __restrict__ gamma,
                         const float* __restrict__ beta) {
  const long long idx = (long long)blockIdx.x * blockDim.x + threadIdx.x;
  if (idx >= total) return;
  const int c = (int)(idx & (GHID - 1));
  const float inv = rsqrtf(stats[GHID + c] + 1e-5f);
  X[idx] = (X[idx] - stats[c]) * inv * gamma[c] + beta[c];
}

// MHA probabilities: block per (b,h) -> awbuf[b,h,n]
__global__ void mha_attn(const float* __restrict__ q,
                         const float* __restrict__ k,
                         float* __restrict__ awbuf) {
  const int b   = blockIdx.x >> 3;
  const int hh  = blockIdx.x & 7;
  const int tid = threadIdx.x;  // 128
  __shared__ float lg[128];
  float v = -INFINITY;
  if (tid < GN) {
    float s = 0.f;
    const float* qp = q + b * GHID + hh * GDH;
    const float* kp = k + (size_t)(b * GN + tid) * GHID + hh * GDH;
#pragma unroll
    for (int d = 0; d < GDH; ++d) s += qp[d] * kp[d];
    v = s * 0.25f;  // DH^-0.5
  }
  lg[tid] = v;
  __syncthreads();
  if (tid == 0) {
    float mx = -INFINITY;
    for (int n = 0; n < GN; ++n) mx = fmaxf(mx, lg[n]);
    float sum = 0.f;
    for (int n = 0; n < GN; ++n) { const float e = __expf(lg[n] - mx); lg[n] = e; sum += e; }
    const float inv = 1.0f / sum;
    for (int n = 0; n < GN; ++n) awbuf[(b * GHEADS + hh) * GN + n] = lg[n] * inv;
  }
}

__global__ void mha_o(const float* __restrict__ awbuf,
                      const float* __restrict__ v, float* __restrict__ o) {
  const int idx = blockIdx.x * blockDim.x + threadIdx.x;  // 4096
  if (idx >= GB * GHID) return;
  const int b = idx >> 7, d = idx & 127, hh = d >> 4;
  float s = 0.f;
  for (int n = 0; n < GN; ++n)
    s += awbuf[(b * GHEADS + hh) * GN + n] * v[(size_t)(b * GN + n) * GHID + d];
  o[idx] = s;
}

__global__ void add_vec(const float* __restrict__ a, const float* __restrict__ b,
                        float* __restrict__ c, int n) {
  const int idx = blockIdx.x * blockDim.x + threadIdx.x;
  if (idx < n) c[idx] = a[idx] + b[idx];
}

__global__ void score_avg(const float* __restrict__ awbuf, float* __restrict__ score) {
  const int idx = blockIdx.x * blockDim.x + threadIdx.x;  // 3200
  if (idx >= GB * GN) return;
  const int b = idx / GN, n = idx % GN;
  float s = 0.f;
#pragma unroll
  for (int hh = 0; hh < GHEADS; ++hh) s += awbuf[(b * GHEADS + hh) * GN + n];
  score[idx] = s * 0.125f;
}

extern "C" void kernel_launch(void* const* d_in, const int* in_sizes, int n_in,
                              void* d_out, int out_size, void* d_ws, size_t ws_size,
                              hipStream_t stream) {
  (void)in_sizes; (void)n_in; (void)out_size; (void)ws_size;
  const float* in_h      = (const float*)d_in[0];
  const int*   adj       = (const int*)d_in[1];
  const float* in_dis    = (const float*)d_in[2];
  const float* in_hstart = (const float*)d_in[3];
  const float* gat_W     = (const float*)d_in[4];
  const float* gat_We    = (const float*)d_in[5];
  const float* gat_a     = (const float*)d_in[6];
  const float* et_Ew     = (const float*)d_in[7];
  const float* et_Eb     = (const float*)d_in[8];
  const float* et_Hw1    = (const float*)d_in[9];
  const float* et_Hb1    = (const float*)d_in[10];
  const float* et_Hw2    = (const float*)d_in[11];
  const float* et_Hb2    = (const float*)d_in[12];
  const float* nbn_g     = (const float*)d_in[13];
  const float* nbn_b     = (const float*)d_in[14];
  const float* ebn_g     = (const float*)d_in[15];
  const float* ebn_b     = (const float*)d_in[16];
  const float* abn_g     = (const float*)d_in[17];
  const float* abn_b     = (const float*)d_in[18];
  const float* mha_in_w  = (const float*)d_in[19];
  const float* mha_in_b  = (const float*)d_in[20];
  const float* mha_out_w = (const float*)d_in[21];
  const float* mha_out_b = (const float*)d_in[22];

  const int    MN        = GB * GN;            // 3200 node rows
  const int    ME        = GB * GN * GN;       // 320000 edge rows
  const size_t DIS_ELEMS = (size_t)ME * GHID;  // 40,960,000
  const int    WSZ       = GHID * GHID;        // 16384

  // ---- workspace carve: f32 region then f16 region ----
  float* ws   = (float*)d_ws;
  float* h    = ws; ws += (size_t)MN * GHID;   // 409600
  float* g    = ws; ws += (size_t)MN * GHID;
  float* he1  = ws; ws += (size_t)MN * GHID;
  float* he2  = ws; ws += (size_t)MN * GHID;
  float* kb   = ws; ws += (size_t)MN * GHID;
  float* vb   = ws; ws += (size_t)MN * GHID;
  float* si   = ws; ws += (size_t)MN * GHEADS;
  float* sj   = ws; ws += (size_t)MN * GHEADS;
  float* awbf = ws; ws += (size_t)GB * GHEADS * GN;
  float* awe  = ws; ws += GHID * GHEADS;
  float* qb   = ws; ws += GB * GHID;
  float* ob   = ws; ws += GB * GHID;
  float* o2   = ws; ws += GB * GHID;
  float* hsA  = ws; ws += GB * GHID;
  float* hsB  = ws; ws += GB * GHID;
  float* stat = ws; ws += 2 * GHID;

  _Float16* hw      = (_Float16*)ws;
  _Float16* gatW16  = hw; hw += 3 * WSZ;
  _Float16* etEw16  = hw; hw += 3 * WSZ;
  _Float16* etHw1h  = hw; hw += 3 * WSZ;
  _Float16* etHw2h  = hw; hw += 3 * WSZ;
  _Float16* mhaInH  = hw; hw += 3 * 3 * WSZ;
  _Float16* mhaOutH = hw; hw += 3 * WSZ;
  _Float16* h16     = hw; hw += (size_t)MN * GHID;
  _Float16* hs16    = hw; hw += GB * GHID;
  _Float16* ob16    = hw; hw += GB * GHID;

  float* dis_out = (float*)d_out;  // working dis == output dis region
  float* score   = (float*)d_out + DIS_ELEMS;

  (void)hipMemcpyAsync(h, in_h, (size_t)MN * GHID * sizeof(float),
                       hipMemcpyDeviceToDevice, stream);
  (void)hipMemcpyAsync(hsA, in_hstart, (size_t)GB * GHID * sizeof(float),
                       hipMemcpyDeviceToDevice, stream);

  auto cvt = [&](const float* src, _Float16* dst, int n) {
    cvt_f2h<<<dim3((unsigned)((n + 255) / 256)), 256, 0, stream>>>(src, dst, n);
  };
  // one-time f16 weight conversion (loop-invariant, removes per-wave v_cvt)
  cvt(gat_W,     gatW16,  3 * WSZ);
  cvt(et_Ew,     etEw16,  3 * WSZ);
  cvt(et_Hw1,    etHw1h,  3 * WSZ);
  cvt(et_Hw2,    etHw2h,  3 * WSZ);
  cvt(mha_in_w,  mhaInH,  3 * 3 * WSZ);
  cvt(mha_out_w, mhaOutH, 3 * WSZ);

  const float* cur_dis = in_dis;  // layer 0 reads the input tensor
  float* hs_cur = hsA;
  float* hs_nxt = hsB;

  auto gemm_grid = [](int M) { return dim3((unsigned)(((M >> 4) + 3) / 4)); };

  for (int i = 0; i < 3; ++i) {
    const float* ga = gat_a + (size_t)i * 48;

    // ---- GAT ----
    cvt(h, h16, MN * GHID);
    gemm_h128_wmma<EPI_NONE, true><<<gemm_grid(MN), 128, 0, stream>>>(
        h16, gatW16 + (size_t)i * WSZ, nullptr, g, MN, nullptr, nullptr, nullptr);
    compute_awe<<<4, 256, 0, stream>>>(gat_We + (size_t)i * WSZ, ga + 32, awe);
    compute_si_sj<<<100, 256, 0, stream>>>(g, ga, si, sj);
    gat_attention<<<MN, 128, 0, stream>>>(cur_dis, adj, awe, si, sj, g, h);
    bn_stats<<<GHID, 256, 0, stream>>>(h, MN, stat);
    bn_apply<<<(unsigned)((size_t)MN * GHID / 256), 256, 0, stream>>>(
        h, (long long)MN * GHID, stat, nbn_g + i * GHID, nbn_b + i * GHID);

    // ---- EdgeTransition (de GEMM fused with sigmoid+residual epilogue) ----
    cvt(h, h16, MN * GHID);  // post-BN activations for Hw1/Hw2/k/v GEMMs
    gemm_h128_wmma<EPI_BIAS, true><<<gemm_grid(MN), 128, 0, stream>>>(
        h16, etHw1h + (size_t)i * WSZ, et_Hb1 + i * GHID, he1, MN,
        nullptr, nullptr, nullptr);
    gemm_h128_wmma<EPI_BIAS, true><<<gemm_grid(MN), 128, 0, stream>>>(
        h16, etHw2h + (size_t)i * WSZ, et_Hb2 + i * GHID, he2, MN,
        nullptr, nullptr, nullptr);
    gemm_h128_wmma<EPI_EDGE, false><<<gemm_grid(ME), 128, 0, stream>>>(
        cur_dis, etEw16 + (size_t)i * WSZ, et_Eb + i * GHID, dis_out, ME,
        he1, he2, cur_dis);
    cur_dis = dis_out;
    bn_stats<<<GHID, 256, 0, stream>>>(dis_out, ME, stat);
    bn_apply<<<(unsigned)(DIS_ELEMS / 256), 256, 0, stream>>>(
        dis_out, (long long)DIS_ELEMS, stat, ebn_g + i * GHID, ebn_b + i * GHID);

    // ---- MultiheadAttention (query = h_start) ----
    const _Float16* Win = mhaInH + (size_t)i * 3 * WSZ;
    const float*    bin = mha_in_b + (size_t)i * 3 * GHID;
    cvt(hs_cur, hs16, GB * GHID);
    gemm_h128_wmma<EPI_BIAS, true><<<gemm_grid(GB), 128, 0, stream>>>(
        hs16, Win, bin, qb, GB, nullptr, nullptr, nullptr);
    gemm_h128_wmma<EPI_BIAS, true><<<gemm_grid(MN), 128, 0, stream>>>(
        h16, Win + WSZ, bin + GHID, kb, MN, nullptr, nullptr, nullptr);
    mha_attn<<<GB * GHEADS, 128, 0, stream>>>(qb, kb, awbf);
    if (i == 2) {
      score_avg<<<(GB * GN + 255) / 256, 256, 0, stream>>>(awbf, score);
    } else {
      gemm_h128_wmma<EPI_BIAS, true><<<gemm_grid(MN), 128, 0, stream>>>(
          h16, Win + 2 * WSZ, bin + 2 * GHID, vb, MN, nullptr, nullptr, nullptr);
      mha_o<<<GB, 128, 0, stream>>>(awbf, vb, ob);
      cvt(ob, ob16, GB * GHID);
      gemm_h128_wmma<EPI_BIAS, true><<<gemm_grid(GB), 128, 0, stream>>>(
          ob16, mhaOutH + (size_t)i * WSZ, mha_out_b + i * GHID, o2, GB,
          nullptr, nullptr, nullptr);
      add_vec<<<16, 256, 0, stream>>>(o2, hs_cur, hs_nxt, GB * GHID);
      bn_stats<<<GHID, 256, 0, stream>>>(hs_nxt, GB, stat);
      bn_apply<<<16, 256, 0, stream>>>(hs_nxt, GB * GHID, stat,
                                       abn_g + i * GHID, abn_b + i * GHID);
      float* t = hs_cur; hs_cur = hs_nxt; hs_nxt = t;
    }
  }
}
